// AttentionBlock_74010876445388
// MI455X (gfx1250) — matrix operand order repaired and checked by reading, not compiled, not run on hardware
//
#include <hip/hip_runtime.h>
#include <hip/hip_bf16.h>

// ---------------------------------------------------------------------------
// AttentionBlock for MI455X (gfx1250): bf16 WMMA GEMMs + fused flash attention
// + double-buffered Tensor Data Mover (TDM) tile staging into LDS.
// B=16, C=512, L=1024, heads=8 (ch=64), groups=32
// ---------------------------------------------------------------------------

typedef __attribute__((ext_vector_type(16))) __bf16 bf16x16;
typedef __attribute__((ext_vector_type(8)))  __bf16 bf16x8;
typedef __attribute__((ext_vector_type(2)))  __bf16 bf16x2;
typedef __attribute__((ext_vector_type(8)))  float  f32x8;
typedef __attribute__((ext_vector_type(4))) unsigned int u32x4;
typedef __attribute__((ext_vector_type(8))) int i32x8;
typedef __attribute__((ext_vector_type(4))) int i32x4;

#define WMMA_BF16(A, B, C) \
  __builtin_amdgcn_wmma_f32_16x16x32_bf16(false, (A), false, (B), (short)0, (C), false, false)

// ---- TDM 2D tile load: global (2-byte elems, row stride in elems) -> LDS ----
// Descriptor packing per CDNA5 ISA.
// pad_icode: LDS padding interval code (4 => every 32 dwords, 5 => every 64)
// pad_acode: padding amount code (3 => 4 dwords)
__device__ __forceinline__ void tdm_load_2d_bf16(unsigned lds_addr, const void* gaddr,
                                                 unsigned tile_w, unsigned tile_h,
                                                 unsigned row_stride_elems,
                                                 unsigned pad_icode, unsigned pad_acode) {
  unsigned long long ga = (unsigned long long)gaddr;
  u32x4 g0;
  g0[0] = 1u;                                   // count=1, user descriptor
  g0[1] = lds_addr;                             // LDS byte address
  g0[2] = (unsigned)(ga & 0xffffffffu);         // global_addr[31:0]
  g0[3] = (unsigned)((ga >> 32) & 0x1ffffffu) | (2u << 30);  // addr[56:32] | type=2
  i32x8 g1;
  unsigned d0 = (1u << 16)                      // data_size = 1 (2 bytes)
              | (1u << 20)                      // pad_enable
              | (pad_icode << 22) | (pad_acode << 25);
  g1[0] = (int)d0;                              // wg_mask=0, no atomic barrier
  g1[1] = (int)((tile_w & 0xffffu) << 16);      // tensor_dim0[15:0] @ bits 63:48
  g1[2] = (int)(((tile_w >> 16) & 0xffffu) | ((tile_h & 0xffffu) << 16)); // dim0 hi | dim1 lo
  g1[3] = (int)(((tile_h >> 16) & 0xffffu) | ((tile_w & 0xffffu) << 16)); // dim1 hi | tile_dim0
  g1[4] = (int)(tile_h & 0xffffu);              // tile_dim1 (tile_dim2 = 0)
  g1[5] = (int)row_stride_elems;                // tensor_dim0_stride[31:0]
  g1[6] = 0;                                    // stride hi / dim1_stride lo
  g1[7] = 0;
  i32x4 z4 = {0, 0, 0, 0};
  i32x8 z8 = {0, 0, 0, 0, 0, 0, 0, 0};
  __builtin_amdgcn_tensor_load_to_lds(g0, g1, z4, z4, z8, 0);
}

__device__ __forceinline__ unsigned lds_off(const void* p) {
  return (unsigned)(unsigned long long)(uintptr_t)p;  // flat addr low 32 = LDS offset
}

// ---- WMMA fragment loaders (layouts per CDNA5 ISA 7.12.2) -------
__device__ __forceinline__ bf16x16 load_a_frag(const __bf16* p0, int rs) {
  int lane = threadIdx.x & 31;
  const __bf16* p = p0 + (lane & 15) * rs + ((lane & 16) ? 8 : 0);
  bf16x16 a;
#pragma unroll
  for (int j = 0; j < 8; ++j) a[j] = p[j];
#pragma unroll
  for (int j = 0; j < 8; ++j) a[8 + j] = p[16 + j];
  return a;
}

__device__ __forceinline__ bf16x16 load_b_frag(const __bf16* p0, int rs) {
  int lane = threadIdx.x & 31;
  const __bf16* p = p0 + (lane & 15) + ((lane & 16) ? 8 : 0) * rs;
  bf16x16 b;
#pragma unroll
  for (int j = 0; j < 8; ++j) b[j] = p[j * rs];
#pragma unroll
  for (int j = 0; j < 8; ++j) b[8 + j] = p[(16 + j) * rs];
  return b;
}

__device__ __forceinline__ bf16x16 load_bt_frag(const __bf16* p0, int rs) {
  int lane = threadIdx.x & 31;
  const __bf16* p = p0 + (lane & 15) * rs + ((lane & 16) ? 8 : 0);
  bf16x16 b;
#pragma unroll
  for (int j = 0; j < 8; ++j) b[j] = p[j];
#pragma unroll
  for (int j = 0; j < 8; ++j) b[8 + j] = p[16 + j];
  return b;
}

// Stage a 128x32 f32 weight tile -> bf16 LDS tile (float2 loads, packed cvt)
__device__ __forceinline__ void stage_a_tile(__bf16 (*dst)[32], const float* gsrc) {
#pragma unroll
  for (int i = 0; i < 8; ++i) {
    int e = (threadIdx.x + i * 256) * 2;
    int r = e >> 5, c = e & 31;
    const float2 w2 = *(const float2*)(gsrc + (size_t)r * 512 + c);
    bf16x2 pk;
    pk[0] = (__bf16)w2.x;
    pk[1] = (__bf16)w2.y;
    *(bf16x2*)&dst[r][c] = pk;
  }
}

// ---------------------------------------------------------------------------
// Kernel 1: GroupNorm (32 groups of 16 ch x 1024) -> bf16 h
// ---------------------------------------------------------------------------
__global__ __launch_bounds__(256) void gn_kernel(const float* __restrict__ x,
                                                 const float* __restrict__ gw,
                                                 const float* __restrict__ gb,
                                                 __bf16* __restrict__ h) {
  int bg = blockIdx.x;
  int b = bg >> 5, g = bg & 31;
  const float* xq = x + (((size_t)(b * 512 + g * 16)) << 10);
  float s = 0.f, ss = 0.f;
  float vals[64];
#pragma unroll
  for (int i = 0; i < 64; ++i) {
    int e = threadIdx.x + i * 256;
    float v0 = xq[e];
    vals[i] = v0;
    s += v0;
    ss += v0 * v0;
  }
#pragma unroll
  for (int m = 1; m < 32; m <<= 1) {
    s += __shfl_xor(s, m, 32);
    ss += __shfl_xor(ss, m, 32);
  }
  __shared__ float red[2][8];
  __shared__ float stats[2];
  int wv = threadIdx.x >> 5, ln = threadIdx.x & 31;
  if (ln == 0) { red[0][wv] = s; red[1][wv] = ss; }
  __syncthreads();
  if (threadIdx.x == 0) {
    float ts = 0.f, tss = 0.f;
#pragma unroll
    for (int i = 0; i < 8; ++i) { ts += red[0][i]; tss += red[1][i]; }
    float mean = ts * (1.f / 16384.f);
    float var = tss * (1.f / 16384.f) - mean * mean;
    stats[0] = mean;
    stats[1] = rsqrtf(var + 1e-5f);
  }
  __syncthreads();
  float mean = stats[0], inv = stats[1];
  __bf16* hp = h + (((size_t)(b * 512 + g * 16)) << 10);
#pragma unroll
  for (int i = 0; i < 64; ++i) {
    int e = threadIdx.x + i * 256;
    int c = g * 16 + (e >> 10);
    float o = (vals[i] - mean) * inv * gw[c] + gb[c];
    hp[e] = (__bf16)o;
  }
}

// ---------------------------------------------------------------------------
// Kernel 2: QKV GEMM  qkv[b,o,l] = sum_c w[o,c] h[b,c,l] + bias[o]
// Double-buffered: TDM prefetches B tile i+1 while WMMAs consume tile i.
// ---------------------------------------------------------------------------
__global__ __launch_bounds__(256) void qkv_kernel(const float* __restrict__ wqkv,
                                                  const float* __restrict__ bqkv,
                                                  const __bf16* __restrict__ h,
                                                  __bf16* __restrict__ qo,
                                                  __bf16* __restrict__ ko,
                                                  __bf16* __restrict__ vo) {
  __shared__ __bf16 As[2][128][32];   // [o][k]  rows 64B
  __shared__ __bf16 Bs[2][32][136];   // [k][l]  rows 272B (TDM pad: 64dw + 4dw)
  int l0 = blockIdx.x * 128;
  int o0 = blockIdx.y * 128;
  int b = blockIdx.z;
  int wv = threadIdx.x >> 5, lane = threadIdx.x & 31;
  const __bf16* hbase = h + (((size_t)b * 512) << 10) + l0;

  f32x8 acc[8];
#pragma unroll
  for (int n = 0; n < 8; ++n)
#pragma unroll
    for (int r = 0; r < 8; ++r) acc[n][r] = 0.f;

  // prologue: tile 0
  if (wv == 0) tdm_load_2d_bf16(lds_off(&Bs[0][0][0]), hbase, 128, 32, 1024, 5, 3);
  stage_a_tile(As[0], wqkv + (size_t)o0 * 512);

  for (int i = 0; i < 16; ++i) {
    int buf = i & 1, nbuf = buf ^ 1;
    __syncthreads();  // all waves done reading buffers [nbuf] from iteration i-1
    if (i + 1 < 16) {
      if (wv == 0)
        tdm_load_2d_bf16(lds_off(&Bs[nbuf][0][0]), hbase + ((size_t)(i + 1) << 15),
                         128, 32, 1024, 5, 3);
      stage_a_tile(As[nbuf], wqkv + (size_t)o0 * 512 + (i + 1) * 32);
    }
    if (wv == 0) {
      if (i + 1 < 16) __builtin_amdgcn_s_wait_tensorcnt(1);
      else            __builtin_amdgcn_s_wait_tensorcnt(0);
    }
    __syncthreads();  // tile i published
    bf16x16 af = load_a_frag(&As[buf][wv * 16][0], 32);
#pragma unroll
    for (int n = 0; n < 8; ++n) {
      bf16x16 bf = load_b_frag(&Bs[buf][0][n * 16], 136);
      acc[n] = WMMA_BF16(af, bf, acc[n]);
    }
  }

  const float scale = 0.35355339059327373f;  // 1/sqrt(sqrt(64))
  int colb = lane & 15;
  int rowadd = (lane >> 4) << 3;
#pragma unroll
  for (int n = 0; n < 8; ++n) {
    int l = l0 + n * 16 + colb;
#pragma unroll
    for (int r = 0; r < 8; ++r) {
      int o = o0 + wv * 16 + r + rowadd;
      float val = acc[n][r] + bqkv[o];
      int hd = o / 192;
      int rem = o - hd * 192;
      size_t bh = (size_t)b * 8 + hd;
      if (rem < 64) {
        qo[((bh * 64 + rem) << 10) + l] = (__bf16)(val * scale);
      } else if (rem < 128) {
        ko[((bh * 64 + (rem - 64)) << 10) + l] = (__bf16)(val * scale);
      } else {
        vo[((bh * 64 + (rem - 128)) << 10) + l] = (__bf16)val;
      }
    }
  }
}

// ---------------------------------------------------------------------------
// Kernel 3: fused flash attention per (head-batch, 64-row t tile).
// K/V tiles (64x64 bf16) double-buffered via TDM (wait tensorcnt<=2 keeps the
// next pair in flight during compute). Output packed as bf16x8 stores.
// ---------------------------------------------------------------------------
__global__ __launch_bounds__(128) void attn_kernel(const __bf16* __restrict__ qi,
                                                   const __bf16* __restrict__ ki,
                                                   const __bf16* __restrict__ vi,
                                                   __bf16* __restrict__ ao) {
  __shared__ __bf16 Qt[64][72];         // [t][c]
  __shared__ __bf16 Ks[2][64][72];      // [c][s]  (TDM pad: 32dw + 4dw)
  __shared__ __bf16 Vs[2][64][72];      // [c][s]
  __shared__ __bf16 Ps[4][16][72];      // per-wave P [t][s]
  int bh = blockIdx.x;
  int t0 = blockIdx.y * 64;
  int wv = threadIdx.x >> 5, lane = threadIdx.x & 31;
  size_t base = (size_t)bh << 16;  // bh*64*1024

  // prefetch K/V tile 0 while Q is staged
  if (wv == 0) {
    tdm_load_2d_bf16(lds_off(&Ks[0][0][0]), ki + base, 64, 64, 1024, 4, 3);
    tdm_load_2d_bf16(lds_off(&Vs[0][0][0]), vi + base, 64, 64, 1024, 4, 3);
  }
  // stage Q transposed: Qt[t][c] = q[bh][c][t0+t]  (paired bf16 loads)
#pragma unroll
  for (int i = 0; i < 16; ++i) {
    int e = (threadIdx.x + i * 128) * 2;
    int c = e >> 6, t = e & 63;
    bf16x2 v2 = *(const bf16x2*)&qi[base + ((size_t)c << 10) + t0 + t];
    Qt[t][c] = v2[0];
    Qt[t + 1][c] = v2[1];
  }
  __syncthreads();
  bf16x16 aq0 = load_a_frag(&Qt[wv * 16][0], 72);
  bf16x16 aq1 = load_a_frag(&Qt[wv * 16][32], 72);

  f32x8 oacc[4];
  float mrun[8], lrun[8];
#pragma unroll
  for (int n = 0; n < 4; ++n)
#pragma unroll
    for (int r = 0; r < 8; ++r) oacc[n][r] = 0.f;
#pragma unroll
  for (int r = 0; r < 8; ++r) { mrun[r] = -1e30f; lrun[r] = 0.f; }

  int rowadd = (lane >> 4) << 3, colb = lane & 15;

  for (int sb = 0; sb < 16; ++sb) {
    int buf = sb & 1, nbuf = buf ^ 1;
    __syncthreads();  // readers of [nbuf] (iteration sb-1) done
    if (wv == 0) {
      if (sb + 1 < 16) {
        size_t nxt = base + (size_t)(sb + 1) * 64;
        tdm_load_2d_bf16(lds_off(&Ks[nbuf][0][0]), ki + nxt, 64, 64, 1024, 4, 3);
        tdm_load_2d_bf16(lds_off(&Vs[nbuf][0][0]), vi + nxt, 64, 64, 1024, 4, 3);
        __builtin_amdgcn_s_wait_tensorcnt(2);
      } else {
        __builtin_amdgcn_s_wait_tensorcnt(0);
      }
    }
    __syncthreads();  // tile sb published

    // S = Q^T K  (16 t-rows x 64 s-cols per wave)
    f32x8 st[4];
#pragma unroll
    for (int n = 0; n < 4; ++n) {
#pragma unroll
      for (int r = 0; r < 8; ++r) st[n][r] = 0.f;
      st[n] = WMMA_BF16(aq0, load_b_frag(&Ks[buf][0][n * 16], 72), st[n]);
      st[n] = WMMA_BF16(aq1, load_b_frag(&Ks[buf][32][n * 16], 72), st[n]);
    }

    // online softmax per row (row = (r, lane-half); 16-lane half reductions)
#pragma unroll
    for (int r = 0; r < 8; ++r) {
      float rm = fmaxf(fmaxf(st[0][r], st[1][r]), fmaxf(st[2][r], st[3][r]));
      rm = fmaxf(rm, __shfl_xor(rm, 1, 32));
      rm = fmaxf(rm, __shfl_xor(rm, 2, 32));
      rm = fmaxf(rm, __shfl_xor(rm, 4, 32));
      rm = fmaxf(rm, __shfl_xor(rm, 8, 32));
      float mnew = fmaxf(mrun[r], rm);
      float corr = __expf(mrun[r] - mnew);
      float rs = 0.f;
#pragma unroll
      for (int n = 0; n < 4; ++n) {
        float pv = __expf(st[n][r] - mnew);
        st[n][r] = pv;
        rs += pv;
      }
      rs += __shfl_xor(rs, 1, 32);
      rs += __shfl_xor(rs, 2, 32);
      rs += __shfl_xor(rs, 4, 32);
      rs += __shfl_xor(rs, 8, 32);
      lrun[r] = lrun[r] * corr + rs;
      mrun[r] = mnew;
#pragma unroll
      for (int n = 0; n < 4; ++n) oacc[n][r] *= corr;
    }

    // P -> LDS (per-wave region) to re-lay out into A fragments
#pragma unroll
    for (int n = 0; n < 4; ++n)
#pragma unroll
      for (int r = 0; r < 8; ++r)
        Ps[wv][r + rowadd][n * 16 + colb] = (__bf16)st[n][r];
    __syncthreads();

    bf16x16 ap0 = load_a_frag(&Ps[wv][0][0], 72);
    bf16x16 ap1 = load_a_frag(&Ps[wv][0][32], 72);
#pragma unroll
    for (int nc = 0; nc < 4; ++nc) {
      oacc[nc] = WMMA_BF16(ap0, load_bt_frag(&Vs[buf][nc * 16][0], 72), oacc[nc]);
      oacc[nc] = WMMA_BF16(ap1, load_bt_frag(&Vs[buf][nc * 16][32], 72), oacc[nc]);
    }
  }

  // epilogue: lane holds 8 consecutive t values for fixed channel c -> b128 store
  int b = bh >> 3, hd = bh & 7;
#pragma unroll
  for (int nc = 0; nc < 4; ++nc) {
    int c = hd * 64 + nc * 16 + colb;
    bf16x8 pk;
#pragma unroll
    for (int r = 0; r < 8; ++r) pk[r] = (__bf16)(oacc[nc][r] / lrun[r]);
    *(bf16x8*)&ao[(((size_t)b * 512 + c) << 10) + t0 + wv * 16 + rowadd] = pk;
  }
}

// ---------------------------------------------------------------------------
// Kernel 4: output projection  out[b,o,l] = sum_c wproj[o,c] a[b,c,l] + bias[o]
// ---------------------------------------------------------------------------
__global__ __launch_bounds__(256) void proj_kernel(const float* __restrict__ wp,
                                                   const float* __restrict__ bp,
                                                   const __bf16* __restrict__ ai,
                                                   float* __restrict__ out) {
  __shared__ __bf16 As[2][128][32];
  __shared__ __bf16 Bs[2][32][136];
  int l0 = blockIdx.x * 128;
  int o0 = blockIdx.y * 128;
  int b = blockIdx.z;
  int wv = threadIdx.x >> 5, lane = threadIdx.x & 31;
  const __bf16* abase = ai + (((size_t)b * 512) << 10) + l0;

  f32x8 acc[8];
#pragma unroll
  for (int n = 0; n < 8; ++n)
#pragma unroll
    for (int r = 0; r < 8; ++r) acc[n][r] = 0.f;

  if (wv == 0) tdm_load_2d_bf16(lds_off(&Bs[0][0][0]), abase, 128, 32, 1024, 5, 3);
  stage_a_tile(As[0], wp + (size_t)o0 * 512);

  for (int i = 0; i < 16; ++i) {
    int buf = i & 1, nbuf = buf ^ 1;
    __syncthreads();
    if (i + 1 < 16) {
      if (wv == 0)
        tdm_load_2d_bf16(lds_off(&Bs[nbuf][0][0]), abase + ((size_t)(i + 1) << 15),
                         128, 32, 1024, 5, 3);
      stage_a_tile(As[nbuf], wp + (size_t)o0 * 512 + (i + 1) * 32);
    }
    if (wv == 0) {
      if (i + 1 < 16) __builtin_amdgcn_s_wait_tensorcnt(1);
      else            __builtin_amdgcn_s_wait_tensorcnt(0);
    }
    __syncthreads();
    bf16x16 af = load_a_frag(&As[buf][wv * 16][0], 32);
#pragma unroll
    for (int n = 0; n < 8; ++n) {
      bf16x16 bf = load_b_frag(&Bs[buf][0][n * 16], 136);
      acc[n] = WMMA_BF16(af, bf, acc[n]);
    }
  }

  int colb = lane & 15;
  int rowadd = (lane >> 4) << 3;
#pragma unroll
  for (int n = 0; n < 8; ++n) {
    int l = l0 + n * 16 + colb;
#pragma unroll
    for (int r = 0; r < 8; ++r) {
      int o = o0 + wv * 16 + r + rowadd;
      out[(((size_t)b * 512 + o) << 10) + l] = acc[n][r] + bp[o];
    }
  }
}

// ---------------------------------------------------------------------------
extern "C" void kernel_launch(void* const* d_in, const int* in_sizes, int n_in,
                              void* d_out, int out_size, void* d_ws, size_t ws_size,
                              hipStream_t stream) {
  const float* x     = (const float*)d_in[0];
  const float* nw    = (const float*)d_in[1];
  const float* nb    = (const float*)d_in[2];
  const float* wqkv  = (const float*)d_in[3];
  const float* bqkv  = (const float*)d_in[4];
  const float* wproj = (const float*)d_in[5];
  const float* bproj = (const float*)d_in[6];
  float* out = (float*)d_out;

  const size_t NE = (size_t)16 * 512 * 1024;  // 8,388,608 elems
  __bf16* h  = (__bf16*)d_ws;
  __bf16* qb = h + NE;
  __bf16* kb = qb + NE;
  __bf16* vb = kb + NE;
  __bf16* ab = vb + NE;

  gn_kernel<<<512, 256, 0, stream>>>(x, nw, nb, h);
  qkv_kernel<<<dim3(8, 12, 16), 256, 0, stream>>>(wqkv, bqkv, h, qb, kb, vb);
  attn_kernel<<<dim3(128, 16), 128, 0, stream>>>(qb, kb, vb, ab);
  proj_kernel<<<dim3(8, 4, 16), 256, 0, stream>>>(wproj, bproj, ab, out);
}
